// DecoderBlockMoE_1649267442345
// MI455X (gfx1250) — matrix operand order repaired and checked by reading, not compiled, not run on hardware
//
#include <hip/hip_runtime.h>

#define DEV __device__ __forceinline__

typedef unsigned short u16;
typedef unsigned int   u32;
typedef unsigned long long u64;
typedef __attribute__((ext_vector_type(16))) __bf16 v16bf;
typedef __attribute__((ext_vector_type(8)))  float  v8f;
typedef __attribute__((ext_vector_type(4)))  u32    su32x4;
typedef __attribute__((ext_vector_type(8)))  u32    su32x8;

// ---------------- problem constants ----------------
constexpr int B_   = 2;
constexpr int S_   = 1024;
constexpr int D_   = 1024;
constexpr int H_   = 16;
constexpr int HD_  = 64;
constexpr int WIN_ = 256;
constexpr int E_   = 8;
constexpr int DFF_ = 4096;
constexpr int T_   = B_ * S_;      // 2048
constexpr int CAP_ = T_ / E_;      // 256
constexpr float EPS_ = 1e-6f;

// ---------------- bf16 helpers ----------------
DEV u16 f2bf(float f) {
  u32 u = __builtin_bit_cast(u32, f);
  u32 r = u + 0x7FFFu + ((u >> 16) & 1u);   // round-to-nearest-even
  return (u16)(r >> 16);
}
DEV float bf2f(u16 h) { return __builtin_bit_cast(float, (u32)h << 16); }
DEV u32 pack2bf(float a, float b) { return (u32)f2bf(a) | ((u32)f2bf(b) << 16); }

// ---------------- LDS offset of a __shared__ object ----------------
DEV unsigned lds_off_of(const void* p) {
#ifdef __HIP_DEVICE_COMPILE__
  return (unsigned)(u64)(__attribute__((address_space(3))) const char*)p;
#else
  (void)p;
  return 0;
#endif
}

// ---------------- Tensor Data Mover: 2D tile load (D# per ISA 8.3/8.4) ----
// data_size field: 0=1B,1=2B,2=4B,3=8B.  Dims/strides in data_size units.
DEV void tdm_load_2d(unsigned lds_byte_off, const void* gptr,
                     unsigned tile_w, unsigned tile_h,
                     u64 row_stride_elems, unsigned ds_field) {
  u64 ga = (u64)(size_t)gptr;
  const unsigned td = 0x40000000u;  // huge tensor dims -> never OOB
  su32x4 g0;
  g0[0] = 1u;                                   // count=1, user mode
  g0[1] = lds_byte_off;                         // lds_addr
  g0[2] = (u32)ga;                              // global_addr[31:0]
  g0[3] = (u32)((ga >> 32) & 0x01FFFFFFull) | (2u << 30);  // addr[56:32] | type=2
  su32x8 g1;
  g1[0] = ds_field << 16;                       // workgroup_mask=0, data_size
  g1[1] = (td & 0xFFFFu) << 16;                 // tensor_dim0[15:0]
  g1[2] = ((td >> 16) & 0xFFFFu) | ((td & 0xFFFFu) << 16);   // dim0 hi | dim1 lo
  g1[3] = ((td >> 16) & 0xFFFFu) | ((tile_w & 0xFFFFu) << 16); // dim1 hi | tile_dim0
  g1[4] = (tile_h & 0xFFFFu);                   // tile_dim1 | tile_dim2=0
  g1[5] = (u32)row_stride_elems;                // tensor_dim0_stride[31:0]
  g1[6] = (u32)((row_stride_elems >> 32) & 0xFFFFull);  // stride hi | dim1_stride lo
  g1[7] = 0u;
  asm volatile("tensor_load_to_lds %0, %1" :: "s"(g0), "s"(g1) : "memory");
}

// ---------------- WMMA helpers ----------------
union FragU { v16bf v; u32 u[8]; u16 s[16]; };

DEV v8f wmma_bf16(v16bf a, v16bf b, v8f c) {
  return __builtin_amdgcn_wmma_f32_16x16x32_bf16(
      false, a, false, b, (short)0, c, false, false);
}

// A-matrix 16x32 bf16 fragment from row-major storage (lda in elements).
DEV v16bf load_a_frag(const u16* base, int lda, int lane) {
  FragU r;
  const u16* row = base + (lane & 15) * lda;
  int hk = (lane >> 4) << 3;
#pragma unroll
  for (int v = 0; v < 8; ++v) {
    int k = ((v & 4) << 2) + hk + ((v & 3) << 1);
    r.u[v] = *(const u32*)(row + k);
  }
  return r.v;
}

// B-matrix 32x16 bf16 fragment from row-major (K x N) storage, ldb elements.
DEV v16bf load_b_frag(const u16* base, int ldb, int lane) {
  FragU r;
  const u16* col = base + (lane & 15);
  int hk = (lane >> 4) << 3;
#pragma unroll
  for (int s = 0; s < 16; ++s) {
    int k = ((s & 8) << 1) + hk + (s & 7);
    r.s[s] = col[k * ldb];
  }
  return r.v;
}

// ---------------- RMSNorm: f32 row -> bf16 row ----------------
__global__ __launch_bounds__(256) void rmsnorm_kernel(
    const float* __restrict__ x, const float* __restrict__ g, u16* __restrict__ out) {
  int t = blockIdx.x;
  const float* row = x + (size_t)t * D_;
  float ss = 0.f;
#pragma unroll
  for (int i = 0; i < D_ / 256; ++i) {
    float v = row[threadIdx.x + i * 256];
    ss += v * v;
  }
#pragma unroll
  for (int off = 16; off >= 1; off >>= 1) ss += __shfl_xor(ss, off);
  __shared__ float red[8];
  __shared__ float scale_s;
  if ((threadIdx.x & 31) == 0) red[threadIdx.x >> 5] = ss;
  __syncthreads();
  if (threadIdx.x == 0) {
    float tot = 0.f;
#pragma unroll
    for (int i = 0; i < 8; ++i) tot += red[i];
    scale_s = rsqrtf(tot / (float)D_ + EPS_);
  }
  __syncthreads();
  float sc = scale_s;
#pragma unroll
  for (int i = 0; i < D_ / 256; ++i) {
    int d = threadIdx.x + i * 256;
    out[(size_t)t * D_ + d] = f2bf(g[d] * row[d] * sc);
  }
}

// ---------------- Generic BF16 WMMA GEMM ----------------
// C[M,N] = A_bf16[M,K] @ B_f32[K,N]  (A staged via TDM, B converted to bf16)
// EPI 0: f32   1: f32 + residual   2: bf16   3: f32 * rowscale[bz*M+m]
template <int EPI>
__global__ __launch_bounds__(256) void gemm_bf16f32(
    const u16* __restrict__ A, const float* __restrict__ B,
    float* __restrict__ C, u16* __restrict__ Cb,
    const float* __restrict__ res, const float* __restrict__ rowscale,
    int M, int N, int K, size_t sA, size_t sB, size_t sC) {
  __shared__ u16 As[128 * 32];
  __shared__ u16 Bs[32 * 128];

  int bz = blockIdx.z;
  A += bz * sA; B += bz * sB;
  int n0 = blockIdx.x * 128, m0 = blockIdx.y * 128;
  int w = threadIdx.x >> 5, lane = threadIdx.x & 31;
  int wm = (w & 1) * 64, wn = (w >> 1) * 32;
  bool w0 = (threadIdx.x < 32);
  unsigned asOff = lds_off_of(As);

  v8f acc[4][2];
#pragma unroll
  for (int mi = 0; mi < 4; ++mi)
#pragma unroll
    for (int ni = 0; ni < 2; ++ni) acc[mi][ni] = v8f{};

  for (int kb = 0; kb < K; kb += 32) {
    __syncthreads();
    if (w0)  // DMA the A tile (128 rows x 32 bf16, row stride K) into LDS
      tdm_load_2d(asOff, A + (size_t)m0 * K + kb, 32, 128, (u64)K, 1);
    {
      const float* Bg = B + (size_t)kb * N + n0;
#pragma unroll
      for (int j = 0; j < 8; ++j) {
        int u = threadIdx.x + 256 * j;
        int r = u >> 6, cp = u & 63;
        float2 f = *(const float2*)(Bg + (size_t)r * N + cp * 2);
        ((u32*)Bs)[u] = pack2bf(f.x, f.y);
      }
      if (kb + 32 < K)
        __builtin_prefetch(B + (size_t)(kb + 32) * N + n0 + (threadIdx.x >> 1), 0, 1);
    }
    if (w0) __builtin_amdgcn_s_wait_tensorcnt(0);
    __syncthreads();

    v16bf af[4], bfrag[2];
#pragma unroll
    for (int mi = 0; mi < 4; ++mi) af[mi] = load_a_frag(As + (wm + mi * 16) * 32, 32, lane);
#pragma unroll
    for (int ni = 0; ni < 2; ++ni) bfrag[ni] = load_b_frag(Bs + wn + ni * 16, 128, lane);
#pragma unroll
    for (int mi = 0; mi < 4; ++mi)
#pragma unroll
      for (int ni = 0; ni < 2; ++ni) acc[mi][ni] = wmma_bf16(af[mi], bfrag[ni], acc[mi][ni]);
  }

  int nl = lane & 15, hk = lane >> 4;
#pragma unroll
  for (int mi = 0; mi < 4; ++mi)
#pragma unroll
    for (int ni = 0; ni < 2; ++ni)
#pragma unroll
      for (int r = 0; r < 8; ++r) {
        int gm = m0 + wm + mi * 16 + r + 8 * hk;
        int gn = n0 + wn + ni * 16 + nl;
        float val = acc[mi][ni][r];
        size_t idx = (size_t)gm * N + gn;
        if (EPI == 0) C[bz * sC + idx] = val;
        if (EPI == 1) C[bz * sC + idx] = val + res[idx];
        if (EPI == 2) Cb[bz * sC + idx] = f2bf(val);
        if (EPI == 3) C[bz * sC + idx] = val * rowscale[bz * M + gm];
      }
}

// ---------------- RoPE + relayout (f32 -> bf16) ----------------
// q,v -> [b,h,s,hd]; k -> TRANSPOSED [b,h,hd,s] so attention B-frags are linear.
__global__ __launch_bounds__(256) void rope_kernel(
    const float* __restrict__ qf, const float* __restrict__ kf, const float* __restrict__ vf,
    u16* __restrict__ qb, u16* __restrict__ kT, u16* __restrict__ vb) {
  size_t idx = (size_t)blockIdx.x * blockDim.x + threadIdx.x;  // T*H*32
  if (idx >= (size_t)T_ * H_ * 32) return;
  int dd = idx & 31;
  size_t th = idx >> 5;
  int h = (int)(th % H_);
  int t = (int)(th / H_);
  int b = t / S_, s = t % S_;
  float inv = __powf(10000.f, -((float)(2 * dd) / (float)HD_));
  float ang = (float)s * inv;
  float c = __cosf(ang), sn = __sinf(ang);
  size_t src = (size_t)t * D_ + h * HD_;
  size_t bh = (size_t)(b * H_ + h);
  size_t dst = (bh * S_ + s) * HD_;
  float q1 = qf[src + dd], q2 = qf[src + dd + 32];
  float k1 = kf[src + dd], k2 = kf[src + dd + 32];
  qb[dst + dd]      = f2bf(q1 * c - q2 * sn);
  qb[dst + dd + 32] = f2bf(q2 * c + q1 * sn);
  kT[(bh * HD_ + dd) * S_ + s]      = f2bf(k1 * c - k2 * sn);
  kT[(bh * HD_ + dd + 32) * S_ + s] = f2bf(k2 * c + k1 * sn);
  vb[dst + dd]      = f2bf(vf[src + dd]);
  vb[dst + dd + 32] = f2bf(vf[src + dd + 32]);
}

// ---------------- Flash attention (WMMA QK^T and PV, TDM double buffer) ----
// grid: (S/64, H, B); 4 waves, each owns 16 queries.
__global__ __launch_bounds__(128) void attn_kernel(
    const u16* __restrict__ qb, const u16* __restrict__ kT, const u16* __restrict__ vb,
    const unsigned char* __restrict__ gtm, u16* __restrict__ attn_out) {
  __shared__ u16 KtT[2][64 * 32];   // [hd][key] tile (transposed K)
  __shared__ u16 Vt[2][32 * 64];    // [key][hd] tile
  __shared__ u16 Pl[4][16 * 32];

  int b = blockIdx.z, h = blockIdx.y;
  int wv = threadIdx.x >> 5, lane = threadIdx.x & 31;
  int q0 = blockIdx.x * 64 + wv * 16;
  int nl = lane & 15, hk = lane >> 4;
  bool w0 = (threadIdx.x < 32);

  const u16* Q = qb + ((size_t)(b * H_ + h) * S_ + q0) * HD_;
  v16bf aq0 = load_a_frag(Q, HD_, lane);
  v16bf aq1 = load_a_frag(Q + 32, HD_, lane);

  int qrow[8];
  int gq[8];
#pragma unroll
  for (int r = 0; r < 8; ++r) {
    qrow[r] = q0 + r + 8 * hk;
    gq[r] = gtm[b * S_ + qrow[r]];
  }

  float mrow[8], lrow[8];
  v8f o[4];
#pragma unroll
  for (int r = 0; r < 8; ++r) { mrow[r] = -INFINITY; lrow[r] = 0.f; }
#pragma unroll
  for (int f = 0; f < 4; ++f) o[f] = v8f{};

  const u16* KTg = kT + (size_t)(b * H_ + h) * HD_ * S_;
  const u16* Vg  = vb + (size_t)(b * H_ + h) * S_ * HD_;
  unsigned koff[2] = {lds_off_of(KtT[0]), lds_off_of(KtT[1])};
  unsigned voff[2] = {lds_off_of(Vt[0]), lds_off_of(Vt[1])};

  if (w0) {
    tdm_load_2d(koff[0], KTg, 32, 64, (u64)S_, 1);          // 64 hd-rows x 32 keys
    tdm_load_2d(voff[0], Vg, 2048, 1, 2048ull, 1);          // contiguous 32x64
  }

  int buf = 0;
  for (int k0 = 0; k0 < S_; k0 += 32) {
    if (w0) __builtin_amdgcn_s_wait_tensorcnt(0);
    __syncthreads();
    if (w0 && (k0 + 32) < S_) {   // prefetch next tile while computing this one
      tdm_load_2d(koff[buf ^ 1], KTg + (k0 + 32), 32, 64, (u64)S_, 1);
      tdm_load_2d(voff[buf ^ 1], Vg + (size_t)(k0 + 32) * HD_, 2048, 1, 2048ull, 1);
    }
    const u16* KTt = KtT[buf];
    const u16* Vtp = Vt[buf];

    // scores: two 16x16 tiles, K-dim 64 = 2 WMMA each (B-frags now linear loads)
    v8f sv[2];
#pragma unroll
    for (int t = 0; t < 2; ++t) {
      v8f c = v8f{};
      c = wmma_bf16(aq0, load_b_frag(KTt + t * 16, 32, lane), c);
      c = wmma_bf16(aq1, load_b_frag(KTt + 32 * 32 + t * 16, 32, lane), c);
      sv[t] = c;
    }

    int gk0 = gtm[b * S_ + k0 + nl];
    int gk1 = gtm[b * S_ + k0 + 16 + nl];

    float pw[2][8];
#pragma unroll
    for (int r = 0; r < 8; ++r) {
      float v0, v1;
      {
        int ka = k0 + nl;
        int ok = ((ka <= qrow[r]) && (qrow[r] - ka < WIN_)) || gq[r] || gk0;
        v0 = ok ? sv[0][r] * 0.125f : -INFINITY;
        ka = k0 + 16 + nl;
        ok = ((ka <= qrow[r]) && (qrow[r] - ka < WIN_)) || gq[r] || gk1;
        v1 = ok ? sv[1][r] * 0.125f : -INFINITY;
      }
      float best = fmaxf(v0, v1);
#pragma unroll
      for (int off = 1; off < 16; off <<= 1) best = fmaxf(best, __shfl_xor(best, off));
      float nm = fmaxf(mrow[r], best);
      float sc = (nm == -INFINITY) ? 1.f : __expf(mrow[r] - nm);
      float p0 = (v0 == -INFINITY) ? 0.f : __expf(v0 - nm);
      float p1 = (v1 == -INFINITY) ? 0.f : __expf(v1 - nm);
      pw[0][r] = p0; pw[1][r] = p1;
      float ps = p0 + p1;
#pragma unroll
      for (int off = 1; off < 16; off <<= 1) ps += __shfl_xor(ps, off);
      lrow[r] = lrow[r] * sc + ps;
      mrow[r] = nm;
#pragma unroll
      for (int f = 0; f < 4; ++f) o[f][r] *= sc;
    }

    // P (16x32 probs) -> LDS bf16 -> A fragment, then PV
    u16* Pw = Pl[wv];
#pragma unroll
    for (int t = 0; t < 2; ++t)
#pragma unroll
      for (int r = 0; r < 8; ++r)
        Pw[(r + 8 * hk) * 32 + t * 16 + nl] = f2bf(pw[t][r]);
    v16bf ap = load_a_frag(Pw, 32, lane);
#pragma unroll
    for (int f = 0; f < 4; ++f)
      o[f] = wmma_bf16(ap, load_b_frag(Vtp + f * 16, 64, lane), o[f]);

    buf ^= 1;
  }

#pragma unroll
  for (int f = 0; f < 4; ++f)
#pragma unroll
    for (int r = 0; r < 8; ++r) {
      float val = o[f][r] / lrow[r];
      attn_out[(size_t)(b * S_ + qrow[r]) * D_ + h * HD_ + f * 16 + nl] = f2bf(val);
    }
}

// ---------------- Gate: softmax over E, top-2, aux accumulation ----------------
__global__ void gate_kernel(const u16* __restrict__ hn, const float* __restrict__ Wg,
                            float* __restrict__ w, float* __restrict__ phi) {
  int t = blockIdx.x, lane = threadIdx.x;
  float acc[E_];
#pragma unroll
  for (int e = 0; e < E_; ++e) acc[e] = 0.f;
  for (int d = lane; d < D_; d += 32) {
    float xv = bf2f(hn[(size_t)t * D_ + d]);
    const float* wr = Wg + (size_t)d * E_;
#pragma unroll
    for (int e = 0; e < E_; ++e) acc[e] += xv * wr[e];
  }
#pragma unroll
  for (int e = 0; e < E_; ++e)
#pragma unroll
    for (int off = 16; off >= 1; off >>= 1) acc[e] += __shfl_xor(acc[e], off);
  if (lane == 0) {
    float mx = acc[0];
#pragma unroll
    for (int e = 1; e < E_; ++e) mx = fmaxf(mx, acc[e]);
    float ex[E_], sum = 0.f;
#pragma unroll
    for (int e = 0; e < E_; ++e) { ex[e] = __expf(acc[e] - mx); sum += ex[e]; }
#pragma unroll
    for (int e = 0; e < E_; ++e) ex[e] /= sum;
    int i1 = 0;
#pragma unroll
    for (int e = 1; e < E_; ++e) if (ex[e] > ex[i1]) i1 = e;
    int i2 = (i1 == 0) ? 1 : 0;
#pragma unroll
    for (int e = 0; e < E_; ++e) if (e != i1 && ex[e] > ex[i2]) i2 = e;
#pragma unroll
    for (int e = 0; e < E_; ++e)
      w[(size_t)t * E_ + e] = (e == i1) ? ex[i1] : ((e == i2) ? ex[i2] : 0.f);
#pragma unroll
    for (int e = 0; e < E_; ++e) atomicAdd(&phi[e], ex[e]);
  }
}

__global__ void aux_kernel(const float* __restrict__ phi, float* __restrict__ out) {
  float a = 0.f;
#pragma unroll
  for (int e = 0; e < E_; ++e) {
    float m = phi[e] / (float)T_;
    a += m * m;
  }
  *out = (float)E_ * a;
}

// ---------------- Per-expert top-CAP selection (tie -> lowest index) ----------------
__global__ __launch_bounds__(256) void expert_select_kernel(
    const float* __restrict__ w, float* __restrict__ selw, int* __restrict__ seli) {
  int e = blockIdx.x, tid = threadIdx.x;
  __shared__ float vals[T_];
  __shared__ float bv[256];
  __shared__ int bi[256];
  for (int i = tid; i < T_; i += 256) vals[i] = w[(size_t)i * E_ + e];
  __syncthreads();
  for (int c = 0; c < CAP_; ++c) {
    float best = -INFINITY;
    int bidx = T_;
    for (int i = tid; i < T_; i += 256) {
      float v = vals[i];
      if (v > best || (v == best && i < bidx)) { best = v; bidx = i; }
    }
    bv[tid] = best; bi[tid] = bidx;
    __syncthreads();
    for (int off = 128; off > 0; off >>= 1) {
      if (tid < off) {
        if (bv[tid + off] > bv[tid] ||
            (bv[tid + off] == bv[tid] && bi[tid + off] < bi[tid])) {
          bv[tid] = bv[tid + off]; bi[tid] = bi[tid + off];
        }
      }
      __syncthreads();
    }
    if (tid == 0) {
      selw[e * CAP_ + c] = bv[0];
      seli[e * CAP_ + c] = bi[0];
      vals[bi[0]] = -INFINITY;
    }
    __syncthreads();
  }
}

// ---------------- gather tokens into (E, CAP, D) bf16 ----------------
__global__ __launch_bounds__(256) void gather_kernel(
    const u16* __restrict__ hn, const int* __restrict__ seli, u16* __restrict__ tok) {
  int ec = blockIdx.x;
  int row = seli[ec];
  u32* dst = (u32*)(tok + (size_t)ec * D_);
  const u32* src = (const u32*)(hn + (size_t)row * D_);
#pragma unroll
  for (int j = 0; j < (D_ / 2) / 256; ++j) dst[threadIdx.x + 256 * j] = src[threadIdx.x + 256 * j];
}

// ---------------- SwiGLU activation: h1 * silu(h2), bf16 -> bf16 ----------------
__global__ __launch_bounds__(256) void act_kernel(const u16* __restrict__ hglu,
                                                  u16* __restrict__ act) {
  size_t idx = (size_t)blockIdx.x * 256 + threadIdx.x;
  if (idx >= (size_t)E_ * CAP_ * DFF_) return;
  size_t ec = idx / DFF_;
  int f = (int)(idx % DFF_);
  const u16* row = hglu + ec * (size_t)(2 * DFF_);
  float h1 = bf2f(row[f]);
  float h2 = bf2f(row[DFF_ + f]);
  float sig = 1.f / (1.f + __expf(-h2));
  act[idx] = f2bf(h1 * (h2 * sig));
}

// ---------------- residual copy + scatter-add ----------------
__global__ __launch_bounds__(256) void copy_kernel(const float* __restrict__ src,
                                                   float* __restrict__ dst) {
  size_t i = (size_t)blockIdx.x * 256 + threadIdx.x;
  if (i < (size_t)T_ * D_) dst[i] = src[i];
}

__global__ __launch_bounds__(256) void scatter_kernel(
    const float* __restrict__ oute, const int* __restrict__ seli, float* __restrict__ out) {
  int ec = blockIdx.x;
  int row = seli[ec];
#pragma unroll
  for (int j = 0; j < D_ / 256; ++j) {
    int d = threadIdx.x + 256 * j;
    atomicAdd(&out[(size_t)row * D_ + d], oute[(size_t)ec * D_ + d]);
  }
}

// ---------------- host orchestration ----------------
extern "C" void kernel_launch(void* const* d_in, const int* in_sizes, int n_in,
                              void* d_out, int out_size, void* d_ws, size_t ws_size,
                              hipStream_t stream) {
  const float* x    = (const float*)d_in[0];
  // d_in[1] = additive causal mask (recomputed analytically in-kernel)
  const unsigned char* gtm = (const unsigned char*)d_in[2];
  const float* Wq   = (const float*)d_in[3];
  const float* Wk   = (const float*)d_in[4];
  const float* Wv   = (const float*)d_in[5];
  const float* Wo   = (const float*)d_in[6];
  const float* g1   = (const float*)d_in[7];
  const float* g2   = (const float*)d_in[8];
  const float* Wg   = (const float*)d_in[9];
  const float* W1   = (const float*)d_in[10];
  const float* W2   = (const float*)d_in[11];
  float* out = (float*)d_out;

  char* ws = (char*)d_ws;
  size_t off = 0;
  auto alloc = [&](size_t bytes) -> char* {
    char* p = ws + off;
    off = (off + bytes + 255) & ~(size_t)255;
    return p;
  };
  u16*   xn   = (u16*)  alloc((size_t)T_ * D_ * 2);
  float* qf   = (float*)alloc((size_t)T_ * D_ * 4);
  float* kf   = (float*)alloc((size_t)T_ * D_ * 4);
  float* vf   = (float*)alloc((size_t)T_ * D_ * 4);
  u16*   qb   = (u16*)  alloc((size_t)T_ * D_ * 2);
  u16*   kTb  = (u16*)  alloc((size_t)T_ * D_ * 2);
  u16*   vbuf = (u16*)  alloc((size_t)T_ * D_ * 2);
  u16*   attn = (u16*)  alloc((size_t)T_ * D_ * 2);
  float* hbuf = (float*)alloc((size_t)T_ * D_ * 4);
  u16*   hn   = (u16*)  alloc((size_t)T_ * D_ * 2);
  float* wTE  = (float*)alloc((size_t)T_ * E_ * 4);
  float* phi  = (float*)alloc(E_ * 4);
  float* selw = (float*)alloc((size_t)E_ * CAP_ * 4);
  int*   seli = (int*)  alloc((size_t)E_ * CAP_ * 4);
  u16*   tok  = (u16*)  alloc((size_t)E_ * CAP_ * D_ * 2);
  u16*   hglu = (u16*)  alloc((size_t)E_ * CAP_ * 2 * DFF_ * 2);
  u16*   actb = (u16*)  alloc((size_t)E_ * CAP_ * DFF_ * 2);
  float* oute = (float*)alloc((size_t)E_ * CAP_ * D_ * 4);
  (void)ws_size; (void)in_sizes; (void)n_in; (void)out_size;

  // 1) xn = rmsnorm(x, g1)
  rmsnorm_kernel<<<T_, 256, 0, stream>>>(x, g1, xn);

  // 2) Q/K/V projections
  dim3 gP(D_ / 128, T_ / 128, 1);
  gemm_bf16f32<0><<<gP, 256, 0, stream>>>(xn, Wq, qf, nullptr, nullptr, nullptr,
                                          T_, D_, D_, 0, 0, 0);
  gemm_bf16f32<0><<<gP, 256, 0, stream>>>(xn, Wk, kf, nullptr, nullptr, nullptr,
                                          T_, D_, D_, 0, 0, 0);
  gemm_bf16f32<0><<<gP, 256, 0, stream>>>(xn, Wv, vf, nullptr, nullptr, nullptr,
                                          T_, D_, D_, 0, 0, 0);

  // 3) RoPE + relayout (K transposed for linear B-fragments)
  {
    size_t tot = (size_t)T_ * H_ * 32;
    rope_kernel<<<(unsigned)((tot + 255) / 256), 256, 0, stream>>>(qf, kf, vf, qb, kTb, vbuf);
  }

  // 4) attention
  attn_kernel<<<dim3(S_ / 64, H_, B_), 128, 0, stream>>>(qb, kTb, vbuf, gtm, attn);

  // 5) h = x + attn @ Wo
  gemm_bf16f32<1><<<gP, 256, 0, stream>>>(attn, Wo, hbuf, nullptr, x, nullptr,
                                          T_, D_, D_, 0, 0, 0);

  // 6) hn = rmsnorm(h, g2)
  rmsnorm_kernel<<<T_, 256, 0, stream>>>(hbuf, g2, hn);

  // 7) gating
  hipMemsetAsync(phi, 0, E_ * sizeof(float), stream);
  gate_kernel<<<T_, 32, 0, stream>>>(hn, Wg, wTE, phi);
  aux_kernel<<<1, 1, 0, stream>>>(phi, out + (size_t)T_ * D_);

  // 8) per-expert routing (top-CAP), gather
  expert_select_kernel<<<E_, 256, 0, stream>>>(wTE, selw, seli);
  gather_kernel<<<E_ * CAP_, 256, 0, stream>>>(hn, seli, tok);

  // 9) expert GEMM1: (E) tok(256x1024) @ W1(1024x8192) -> bf16
  gemm_bf16f32<2><<<dim3(2 * DFF_ / 128, CAP_ / 128, E_), 256, 0, stream>>>(
      tok, W1, nullptr, hglu, nullptr, nullptr,
      CAP_, 2 * DFF_, D_,
      (size_t)CAP_ * D_, (size_t)D_ * 2 * DFF_, (size_t)CAP_ * 2 * DFF_);

  // 10) SwiGLU
  {
    size_t tot = (size_t)E_ * CAP_ * DFF_;
    act_kernel<<<(unsigned)((tot + 255) / 256), 256, 0, stream>>>(hglu, actb);
  }

  // 11) expert GEMM2 with per-row sel_w scaling
  gemm_bf16f32<3><<<dim3(D_ / 128, CAP_ / 128, E_), 256, 0, stream>>>(
      actb, W2, oute, nullptr, nullptr, selw,
      CAP_, D_, DFF_,
      (size_t)CAP_ * DFF_, (size_t)DFF_ * D_, (size_t)CAP_ * D_);

  // 12) out = h ; out[sel_idx] += out_e
  copy_kernel<<<(T_ * D_ + 255) / 256, 256, 0, stream>>>(hbuf, out);
  scatter_kernel<<<E_ * CAP_, 256, 0, stream>>>(oute, seli, out);
}
